// LoRAMultiheadAttention_76373108457660
// MI455X (gfx1250) — matrix operand order repaired and checked by reading, not compiled, not run on hardware
//
#include <hip/hip_runtime.h>

// ---------------------------------------------------------------- types
typedef __attribute__((ext_vector_type(16))) __bf16        v16bf;
typedef __attribute__((ext_vector_type(8)))  __bf16        v8bf;
typedef __attribute__((ext_vector_type(8)))  float         v8f;
typedef __attribute__((ext_vector_type(8)))  unsigned short v8u16;

#define EMB    1024
#define SEQ    2048
#define BATCH  2
#define MROWS  (SEQ * BATCH)   /* 4096 */
#define NHEADS 32              /* BATCH * 16 */
#define HDIM   64
#define SCALING 0.25f          /* 1 / rank */

// fp32 -> bf16, round-to-nearest-even
static __device__ __forceinline__ unsigned short f32_to_bf16(float f) {
  unsigned int u = __float_as_uint(f);
  u += 0x7FFFu + ((u >> 16) & 1u);
  return (unsigned short)(u >> 16);
}

static __device__ __forceinline__ v16bf make16(v8u16 lo, v8u16 hi) {
  v8bf a = __builtin_bit_cast(v8bf, lo);
  v8bf b = __builtin_bit_cast(v8bf, hi);
  return __builtin_shufflevector(a, b, 0,1,2,3,4,5,6,7,8,9,10,11,12,13,14,15);
}

// A fragment: 16(M) x 32(K) bf16 from row-major [ld] memory.
// ISA layout: lane = m + 16*half; element i (i<8): K = half*8 + i ;
//             element i (i>=8):   K = 16 + half*8 + (i-8)
static __device__ __forceinline__ v16bf load_a(const unsigned short* base, int ld,
                                               int m0, int k0) {
  int lane = threadIdx.x & 31;
  int m = lane & 15, hf = lane >> 4;
  const unsigned short* p = base + (size_t)(m0 + m) * ld + k0 + hf * 8;
  v8u16 lo = *(const v8u16*)p;         // K = hf*8 .. +7
  v8u16 hi = *(const v8u16*)(p + 16);  // K = 16 + hf*8 .. +7
  return make16(lo, hi);
}

// B fragment: 32(K) x 16(N) bf16 where memory is row-major [n][k] (i.e. B^T).
// ISA layout: lane = n + 16*kh; element i (i<8): K = kh*8 + i ; element i (i>=8): K = 16 + kh*8 + (i-8).
static __device__ __forceinline__ v16bf load_b(const unsigned short* base, int ld,
                                               int n0, int k0) {
  int lane = threadIdx.x & 31;
  int n = lane & 15, kh = lane >> 4;
  const unsigned short* p = base + (size_t)(n0 + n) * ld + k0 + kh * 8;
  v8u16 lo = *(const v8u16*)p;
  v8u16 hi = *(const v8u16*)(p + 16);
  return make16(lo, hi);
}

static __device__ __forceinline__ v8f wmma_bf16(v16bf a, v16bf b, v8f c) {
  return __builtin_amdgcn_wmma_f32_16x16x32_bf16(false, a, false, b,
                                                 (short)0, c, false, false);
}

// Async global->LDS copy of 16 bytes per lane (CDNA5 ASYNCcnt path, ISA §10 async).
// vdst slot carries the LDS byte offset; "off" = no SADDR.
static __device__ __forceinline__ void async_copy_b128(unsigned short* lds_dst,
                                                       const unsigned short* gsrc) {
  unsigned lds_off = (unsigned)(unsigned long long)lds_dst;  // low 32 bits = LDS offset
  unsigned long long ga = (unsigned long long)gsrc;
  asm volatile("global_load_async_to_lds_b128 %0, %1, off"
               :: "v"(lds_off), "v"(ga) : "memory");
}

static __device__ __forceinline__ void wait_asynccnt0() {
  asm volatile("s_wait_asynccnt 0x0" ::: "memory");
}

// ---------------------------------------------------------------- convert
__global__ void cvt_bf16_kernel(const float* __restrict__ in,
                                unsigned short* __restrict__ out, int n) {
  int i = blockIdx.x * blockDim.x + threadIdx.x;
  int stride = gridDim.x * blockDim.x;
  for (; i < n; i += stride) out[i] = f32_to_bf16(in[i]);
}

// ---------------------------------------------------------------- t = x @ A^T  (MROWS x 4)
__global__ void lora_t_kernel(const float* __restrict__ x,
                              const float* __restrict__ A,
                              float* __restrict__ t) {
  __shared__ float red[4][256];
  int row = blockIdx.x;
  int tid = threadIdx.x;
  float a0 = 0.f, a1 = 0.f, a2 = 0.f, a3 = 0.f;
  const float* xr = x + (size_t)row * EMB;
  for (int i = tid; i < EMB; i += 256) {
    float xv = xr[i];
    a0 += xv * A[i];
    a1 += xv * A[EMB + i];
    a2 += xv * A[2 * EMB + i];
    a3 += xv * A[3 * EMB + i];
  }
  red[0][tid] = a0; red[1][tid] = a1; red[2][tid] = a2; red[3][tid] = a3;
  __syncthreads();
  for (int s = 128; s > 0; s >>= 1) {
    if (tid < s) {
      red[0][tid] += red[0][tid + s];
      red[1][tid] += red[1][tid + s];
      red[2][tid] += red[2][tid + s];
      red[3][tid] += red[3][tid + s];
    }
    __syncthreads();
  }
  if (tid < 4) t[row * 4 + tid] = red[tid][0];
}

// ---------------------------------------------------------------- projection GEMM + LoRA
__global__ void proj_gemm_kernel(const unsigned short* __restrict__ Xbf,
                                 const unsigned short* __restrict__ Wbf,
                                 const float* __restrict__ tmat,
                                 const float* __restrict__ Bm,
                                 unsigned short* __restrict__ outHead,  // (head, l, d) or null
                                 unsigned short* __restrict__ outVt) {  // (head, d, l) or null
  int wid = blockIdx.x * (blockDim.x >> 5) + (threadIdx.x >> 5);
  int mt = wid >> 4;       // 0..255  : 16-row tile
  int ng = wid & 15;       // 0..15   : 64-col group
  int lane = threadIdx.x & 31;
  int n = lane & 15, hf = lane >> 4;
  int m0 = mt * 16;
  int n0 = ng * 64;

  v8f acc0 = {}, acc1 = {}, acc2 = {}, acc3 = {};
  for (int k0 = 0; k0 < EMB; k0 += 32) {
    if (k0 + 32 < EMB)
      __builtin_prefetch(Xbf + (size_t)m0 * EMB + k0 + 32, 0, 0);
    v16bf a = load_a(Xbf, EMB, m0, k0);
    acc0 = wmma_bf16(a, load_b(Wbf, EMB, n0 +  0, k0), acc0);
    acc1 = wmma_bf16(a, load_b(Wbf, EMB, n0 + 16, k0), acc1);
    acc2 = wmma_bf16(a, load_b(Wbf, EMB, n0 + 32, k0), acc2);
    acc3 = wmma_bf16(a, load_b(Wbf, EMB, n0 + 48, k0), acc3);
  }
  v8f accs[4] = {acc0, acc1, acc2, acc3};

#pragma unroll
  for (int st = 0; st < 4; ++st) {
    int o = n0 + st * 16 + n;
    float b0 = Bm[o * 4 + 0], b1 = Bm[o * 4 + 1];
    float b2 = Bm[o * 4 + 2], b3 = Bm[o * 4 + 3];
    int h = o >> 6, d = o & 63;
#pragma unroll
    for (int v = 0; v < 8; ++v) {
      int m = m0 + v + 8 * hf;               // row = l*2 + b
      const float* tr = tmat + m * 4;
      float val = accs[st][v] +
                  SCALING * (tr[0] * b0 + tr[1] * b1 + tr[2] * b2 + tr[3] * b3);
      int l = m >> 1, b = m & 1;
      int head = b * 16 + h;
      unsigned short bv = f32_to_bf16(val);
      if (outHead) outHead[((size_t)head * SEQ + l) * HDIM + d] = bv;
      if (outVt)   outVt[((size_t)head * HDIM + d) * SEQ + l]   = bv;
    }
  }
}

// ---------------------------------------------------------------- flash attention
// 8 waves/block share one head; K/V tiles double-buffered in LDS via async copies.
__global__ void attn_kernel(const unsigned short* __restrict__ qh,
                            const unsigned short* __restrict__ kh,
                            const unsigned short* __restrict__ vt,
                            unsigned short* __restrict__ attn) {
  __shared__ __align__(16) unsigned short Ks[2][32][64];   // 8 KB: 32 keys x 64 d
  __shared__ __align__(16) unsigned short Vs[2][64][32];   // 8 KB: 64 d x 32 keys
  __shared__ __align__(16) unsigned short Pld[8][16][32];  // 8 KB: per-wave P slab

  int tid = threadIdx.x;
  int w = tid >> 5;
  int head = blockIdx.x >> 4;   // 0..31 (all waves in block share head)
  int qg = blockIdx.x & 15;     // 16 groups of 8 q-tiles
  int qt = qg * 8 + w;
  int lane = tid & 31;
  int n = lane & 15, hf = lane >> 4;

  const unsigned short* qb = qh + (size_t)head * SEQ * HDIM;
  const unsigned short* kb = kh + (size_t)head * SEQ * HDIM;
  const unsigned short* vb = vt + (size_t)head * HDIM * SEQ;

  // cooperative async stage of one 32-key K tile + V tile (8 KB total, 2 ops/thread)
  int krow = tid >> 3, kseg = tid & 7;   // K: 32 rows x 64 -> 8x16B per row
  int vrow = tid >> 2, vseg = tid & 3;   // V: 64 rows x 32 -> 4x16B per row
  auto stage = [&](int buf, int kk) {
    async_copy_b128(&Ks[buf][krow][kseg * 8],
                    kb + (size_t)(kk + krow) * HDIM + kseg * 8);
    async_copy_b128(&Vs[buf][vrow][vseg * 8],
                    vb + (size_t)vrow * SEQ + kk + vseg * 8);
  };

  v16bf aq0 = load_a(qb, HDIM, qt * 16, 0);
  v16bf aq1 = load_a(qb, HDIM, qt * 16, 32);

  float mi[8], li[8];
  v8f o0 = {}, o1 = {}, o2 = {}, o3 = {};
#pragma unroll
  for (int v = 0; v < 8; ++v) { mi[v] = -1e30f; li[v] = 0.f; }

  stage(0, 0);
  wait_asynccnt0();
  __syncthreads();

  int buf = 0;
  for (int kk = 0; kk < SEQ; kk += 32) {
    if (kk + 32 < SEQ) stage(buf ^ 1, kk + 32);   // prefetch next tile (ASYNCcnt)

    const unsigned short* kTile = &Ks[buf][0][0];
    const unsigned short* vTile = &Vs[buf][0][0];

    // S tiles (16 queries x 32 keys) from LDS
    v8f s0 = {}, s1 = {};
    s0 = wmma_bf16(aq0, load_b(kTile, HDIM, 0,  0),  s0);
    s0 = wmma_bf16(aq1, load_b(kTile, HDIM, 0,  32), s0);
    s1 = wmma_bf16(aq0, load_b(kTile, HDIM, 16, 0),  s1);
    s1 = wmma_bf16(aq1, load_b(kTile, HDIM, 16, 32), s1);
    s0 = s0 * 0.125f;   // 1/sqrt(64)
    s1 = s1 * 0.125f;

    float al[8];
#pragma unroll
    for (int v = 0; v < 8; ++v) {
      float mx = fmaxf(s0[v], s1[v]);
#pragma unroll
      for (int off = 1; off < 16; off <<= 1)
        mx = fmaxf(mx, __shfl_xor(mx, off, 32));   // reduce over n within half
      float mn = fmaxf(mi[v], mx);
      al[v] = __expf(mi[v] - mn);
      float p0 = __expf(s0[v] - mn);
      float p1 = __expf(s1[v] - mn);
      float rs = p0 + p1;
#pragma unroll
      for (int off = 1; off < 16; off <<= 1)
        rs += __shfl_xor(rs, off, 32);
      li[v] = li[v] * al[v] + rs;
      mi[v] = mn;
      s0[v] = p0; s1[v] = p1;
      o0[v] *= al[v]; o1[v] *= al[v]; o2[v] *= al[v]; o3[v] *= al[v];
    }

    // C-layout -> A-layout through per-wave LDS slab
#pragma unroll
    for (int v = 0; v < 8; ++v) {
      Pld[w][v + 8 * hf][n]      = f32_to_bf16(s0[v]);
      Pld[w][v + 8 * hf][16 + n] = f32_to_bf16(s1[v]);
    }
    __syncthreads();
    v16bf ap = load_a(&Pld[w][0][0], 32, 0, 0);
    __syncthreads();

    // O += P @ V  (V tile: row = d, 32 contiguous keys)
    o0 = wmma_bf16(ap, load_b(vTile, 32, 0,  0), o0);
    o1 = wmma_bf16(ap, load_b(vTile, 32, 16, 0), o1);
    o2 = wmma_bf16(ap, load_b(vTile, 32, 32, 0), o2);
    o3 = wmma_bf16(ap, load_b(vTile, 32, 48, 0), o3);

    wait_asynccnt0();     // drain this iteration's prefetch before flipping
    __syncthreads();
    buf ^= 1;
  }

  int b = head >> 4, h = head & 15;
  v8f outs[4] = {o0, o1, o2, o3};
#pragma unroll
  for (int st = 0; st < 4; ++st) {
#pragma unroll
    for (int v = 0; v < 8; ++v) {
      int m = v + 8 * hf;
      int l = qt * 16 + m;
      float val = outs[st][v] / li[v];
      attn[((size_t)(l * 2 + b)) * EMB + h * 64 + st * 16 + n] = f32_to_bf16(val);
    }
  }
}

// ---------------------------------------------------------------- output GEMM + bias (fp32 out)
__global__ void out_gemm_kernel(const unsigned short* __restrict__ Abf,
                                const unsigned short* __restrict__ Wbf,
                                const float* __restrict__ bias,
                                float* __restrict__ out) {
  int wid = blockIdx.x * (blockDim.x >> 5) + (threadIdx.x >> 5);
  int mt = wid >> 4;
  int ng = wid & 15;
  int lane = threadIdx.x & 31;
  int n = lane & 15, hf = lane >> 4;
  int m0 = mt * 16;
  int n0 = ng * 64;

  v8f acc0 = {}, acc1 = {}, acc2 = {}, acc3 = {};
  for (int k0 = 0; k0 < EMB; k0 += 32) {
    if (k0 + 32 < EMB)
      __builtin_prefetch(Abf + (size_t)m0 * EMB + k0 + 32, 0, 0);
    v16bf a = load_a(Abf, EMB, m0, k0);
    acc0 = wmma_bf16(a, load_b(Wbf, EMB, n0 +  0, k0), acc0);
    acc1 = wmma_bf16(a, load_b(Wbf, EMB, n0 + 16, k0), acc1);
    acc2 = wmma_bf16(a, load_b(Wbf, EMB, n0 + 32, k0), acc2);
    acc3 = wmma_bf16(a, load_b(Wbf, EMB, n0 + 48, k0), acc3);
  }
  v8f accs[4] = {acc0, acc1, acc2, acc3};

#pragma unroll
  for (int st = 0; st < 4; ++st) {
    int o = n0 + st * 16 + n;
    float bv = bias[o];
#pragma unroll
    for (int v = 0; v < 8; ++v) {
      int m = m0 + v + 8 * hf;
      out[(size_t)m * EMB + o] = accs[st][v] + bv;
    }
  }
}

// ---------------------------------------------------------------- launch
extern "C" void kernel_launch(void* const* d_in, const int* in_sizes, int n_in,
                              void* d_out, int out_size, void* d_ws, size_t ws_size,
                              hipStream_t stream) {
  (void)in_sizes; (void)n_in; (void)out_size; (void)ws_size;
  const float* query  = (const float*)d_in[0];
  const float* key_in = (const float*)d_in[1];
  const float* value  = (const float*)d_in[2];
  const float* Wq = (const float*)d_in[3];
  const float* Aq = (const float*)d_in[4];
  const float* Bq = (const float*)d_in[5];
  const float* Wk = (const float*)d_in[6];
  const float* Ak = (const float*)d_in[7];
  const float* Bk = (const float*)d_in[8];
  const float* Wv = (const float*)d_in[9];
  const float* Av = (const float*)d_in[10];
  const float* Bv = (const float*)d_in[11];
  const float* Wo = (const float*)d_in[12];
  const float* bo = (const float*)d_in[13];
  float* out = (float*)d_out;

  char* ws = (char*)d_ws;
  size_t off = 0;
  auto alloc = [&](size_t bytes) -> char* {
    char* p = ws + off;
    off = (off + bytes + 255) & ~(size_t)255;
    return p;
  };

  const size_t WEL = (size_t)EMB * EMB;      // 1M
  const size_t XEL = (size_t)MROWS * EMB;    // 4M
  unsigned short* wq_bf = (unsigned short*)alloc(WEL * 2);
  unsigned short* wk_bf = (unsigned short*)alloc(WEL * 2);
  unsigned short* wv_bf = (unsigned short*)alloc(WEL * 2);
  unsigned short* wo_bf = (unsigned short*)alloc(WEL * 2);
  unsigned short* xq_bf = (unsigned short*)alloc(XEL * 2);
  unsigned short* xk_bf = (unsigned short*)alloc(XEL * 2);
  unsigned short* xv_bf = (unsigned short*)alloc(XEL * 2);
  float* tq = (float*)alloc((size_t)MROWS * 4 * 4);
  float* tk = (float*)alloc((size_t)MROWS * 4 * 4);
  float* tv = (float*)alloc((size_t)MROWS * 4 * 4);
  unsigned short* qhm = (unsigned short*)alloc(XEL * 2);  // (head, l, d)
  unsigned short* khm = (unsigned short*)alloc(XEL * 2);  // (head, l, d)
  unsigned short* vtm = (unsigned short*)alloc(XEL * 2);  // (head, d, l)
  unsigned short* attn_bf = (unsigned short*)alloc(XEL * 2);

  // 1) fp32 -> bf16
  cvt_bf16_kernel<<<2048, 256, 0, stream>>>(query,  xq_bf, (int)XEL);
  cvt_bf16_kernel<<<2048, 256, 0, stream>>>(key_in, xk_bf, (int)XEL);
  cvt_bf16_kernel<<<2048, 256, 0, stream>>>(value,  xv_bf, (int)XEL);
  cvt_bf16_kernel<<<1024, 256, 0, stream>>>(Wq, wq_bf, (int)WEL);
  cvt_bf16_kernel<<<1024, 256, 0, stream>>>(Wk, wk_bf, (int)WEL);
  cvt_bf16_kernel<<<1024, 256, 0, stream>>>(Wv, wv_bf, (int)WEL);
  cvt_bf16_kernel<<<1024, 256, 0, stream>>>(Wo, wo_bf, (int)WEL);

  // 2) LoRA down-projections (fp32; tiny)
  lora_t_kernel<<<MROWS, 256, 0, stream>>>(query,  Aq, tq);
  lora_t_kernel<<<MROWS, 256, 0, stream>>>(key_in, Ak, tk);
  lora_t_kernel<<<MROWS, 256, 0, stream>>>(value,  Av, tv);

  // 3) q/k/v projections (WMMA), scattering into attention layouts
  proj_gemm_kernel<<<512, 256, 0, stream>>>(xq_bf, wq_bf, tq, Bq, qhm, nullptr);
  proj_gemm_kernel<<<512, 256, 0, stream>>>(xk_bf, wk_bf, tk, Bk, khm, nullptr);
  proj_gemm_kernel<<<512, 256, 0, stream>>>(xv_bf, wv_bf, tv, Bv, nullptr, vtm);

  // 4) attention (WMMA flash, online softmax, async double-buffered K/V tiles)
  attn_kernel<<<512, 256, 0, stream>>>(qhm, khm, vtm, attn_bf);

  // 5) output projection + bias -> fp32
  out_gemm_kernel<<<512, 256, 0, stream>>>(attn_bf, wo_bf, bo, out);
}
